// Recurrent_model_61151744361114
// MI455X (gfx1250) — compile-verified, matrix operand order checked by reading, not run on hardware
//
#include <hip/hip_runtime.h>
#include <hip/hip_bf16.h>
#include <math.h>

// Problem constants (match reference).
#define B_SZ   2048
#define T_SZ   12
#define IMG    512
#define HID    1024
#define NCLS   51
#define M_ROWS (T_SZ * B_SZ)   // 24576 GEMM rows, (t,b) ordering

typedef __attribute__((ext_vector_type(16))) __bf16 v16bf;
typedef __attribute__((ext_vector_type(8)))  float  v8f;

__device__ __forceinline__ unsigned short f2bf(float f) {
    unsigned int u = __float_as_uint(f);
    u += 0x7FFFu + ((u >> 16) & 1u);          // round-to-nearest-even
    return (unsigned short)(u >> 16);
}
__device__ __forceinline__ float bf2f(unsigned short h) {
    return __uint_as_float(((unsigned int)h) << 16);
}

// ---------------------------------------------------------------------------
// Reorder x (B,T,IMG) f32 -> Xbf (T*B, IMG) bf16, row m = t*B + b.
// ---------------------------------------------------------------------------
__global__ void cvt_x_kernel(const float* __restrict__ x,
                             unsigned short* __restrict__ xbf) {
    size_t i = (size_t)blockIdx.x * blockDim.x + threadIdx.x;  // < M_ROWS*IMG
    size_t row = i >> 9;               // / IMG (512)
    int    col = (int)(i & (IMG - 1));
    int t = (int)(row / B_SZ);
    int b = (int)(row % B_SZ);
    xbf[i] = f2bf(x[((size_t)b * T_SZ + (size_t)t) * IMG + col]);
}

// ---------------------------------------------------------------------------
// W (K,N) f32 row-major -> Wt (N,K) bf16 row-major (so B-fragment loads are
// contiguous, mirroring the A-matrix LDS access pattern).
// ---------------------------------------------------------------------------
__global__ void cvt_w_kernel(const float* __restrict__ W,
                             unsigned short* __restrict__ Wt,
                             int K, int N) {
    size_t i = (size_t)blockIdx.x * blockDim.x + threadIdx.x;  // < (size_t)N*K
    int n = (int)(i / (size_t)K);
    int k = (int)(i - (size_t)n * K);
    Wt[i] = f2bf(W[(size_t)k * N + n]);
}

// ---------------------------------------------------------------------------
// C[M,N] f32 = A[M,K] (bf16 row-major) x Wt[N,K]^T (bf16 row-major).
// 8 waves (wave32) per WG; WG tile 128x128; wave tile 32x64 (2x4 WMMA tiles).
// ---------------------------------------------------------------------------
__global__ __launch_bounds__(256)
void gemm_bf16_kernel(const unsigned short* __restrict__ A,
                      const unsigned short* __restrict__ Bm,
                      float* __restrict__ C,
                      int M, int N, int K)
{
    __shared__ __align__(16) unsigned short lsA[128 * 32];  // 8 KB
    __shared__ __align__(16) unsigned short lsB[128 * 32];  // 8 KB

    const int tid  = threadIdx.x;
    const int wave = tid >> 5;       // 0..7
    const int lane = tid & 31;
    const int r    = lane & 15;      // row-in-tile / N-in-tile
    const int half = lane >> 4;      // K-block selector per ISA layout
    const int wm   = wave >> 1;      // 0..3  -> M offset wm*32
    const int wn   = wave & 1;       // 0..1  -> N offset wn*64

    const int m0 = blockIdx.y * 128;
    const int n0 = blockIdx.x * 128;

    v8f acc[2][4];
    #pragma unroll
    for (int i = 0; i < 2; ++i)
        #pragma unroll
        for (int j = 0; j < 4; ++j)
            #pragma unroll
            for (int e = 0; e < 8; ++e) acc[i][j][e] = 0.0f;

    union Frag { uint4 u[2]; v16bf v; };

    for (int k0 = 0; k0 < K; k0 += 32) {
        __syncthreads();
        // Cooperative stage: 128 rows x 32 bf16 (64 B/row) for A and B.
        // 512 16-byte chunks each; 256 threads x 2 chunks.
        #pragma unroll
        for (int c = 0; c < 2; ++c) {
            int chunk = tid + c * 256;
            int row = chunk >> 2;        // 0..127
            int c4  = chunk & 3;         // 16B chunk within row
            ((uint4*)lsA)[chunk] =
                *(const uint4*)(A + ((size_t)(m0 + row)) * K + k0 + c4 * 8);
            ((uint4*)lsB)[chunk] =
                *(const uint4*)(Bm + ((size_t)(n0 + row)) * K + k0 + c4 * 8);
        }
        __syncthreads();

        // Fragment loads per ISA 16-bit operand layout:
        // lanes 0-15: K 0..7 (VGPR0-3) and 16..23 (VGPR4-7)
        // lanes 16-31: K 8..15 and 24..31  -> base offset half*8 elems, +16 elems.
        v16bf afrag[2], bfrag[4];
        #pragma unroll
        for (int mt = 0; mt < 2; ++mt) {
            int rowm = wm * 32 + mt * 16 + r;
            const uint4* p = (const uint4*)(lsA + rowm * 32 + half * 8);
            Frag f; f.u[0] = p[0]; f.u[1] = p[2];
            afrag[mt] = f.v;
        }
        #pragma unroll
        for (int nt = 0; nt < 4; ++nt) {
            int rown = wn * 64 + nt * 16 + r;
            const uint4* p = (const uint4*)(lsB + rown * 32 + half * 8);
            Frag f; f.u[0] = p[0]; f.u[1] = p[2];
            bfrag[nt] = f.v;
        }

        #pragma unroll
        for (int mt = 0; mt < 2; ++mt)
            #pragma unroll
            for (int nt = 0; nt < 4; ++nt)
                acc[mt][nt] = __builtin_amdgcn_wmma_f32_16x16x32_bf16(
                    false, afrag[mt], false, bfrag[nt],
                    (short)0, acc[mt][nt], false, false);
    }

    // C/D layout: VGPR v -> M = v + 8*half, N = r (within 16x16 tile).
    #pragma unroll
    for (int mt = 0; mt < 2; ++mt) {
        #pragma unroll
        for (int nt = 0; nt < 4; ++nt) {
            int mrow = m0 + wm * 32 + mt * 16 + 8 * half;
            int ncol = n0 + wn * 64 + nt * 16 + r;
            #pragma unroll
            for (int v = 0; v < 8; ++v)
                C[((size_t)(mrow + v)) * N + ncol] = acc[mt][nt][v];
        }
    }
}

// ---------------------------------------------------------------------------
// SRU recurrence over T for one (b,h). U rows are (t*B + b), ncols = k*HID,
// u_j at column j*HID + h. resid: from U col 3*HID (layer 0) or prev-layer h.
// Writes next-layer bf16 input and/or the T-mean (for the final layer).
// ---------------------------------------------------------------------------
__global__ void sru_scan_kernel(const float* __restrict__ U, int ncols,
                                const unsigned short* __restrict__ residbf,
                                const float* __restrict__ vc,
                                const float* __restrict__ bias,
                                unsigned short* __restrict__ hout,
                                float* __restrict__ hbar)
{
    int g = blockIdx.x * blockDim.x + threadIdx.x;   // b*HID + h
    int b = g >> 10;                                  // / HID
    int h = g & (HID - 1);
    const float vf  = vc[h];
    const float vr  = vc[HID + h];
    const float bf_ = bias[h];
    const float br  = bias[HID + h];
    const float SCALE_X = 1.7320508075688772f;        // sqrt(1 + e^0 * 2)

    float c = 0.0f, hsum = 0.0f;
    for (int t = 0; t < T_SZ; ++t) {
        size_t row = (size_t)(t * B_SZ + b);
        size_t ro  = row * (size_t)ncols;
        float u0 = U[ro + h];
        float u1 = U[ro + HID + h];
        float u2 = U[ro + 2 * HID + h];
        float xr = residbf ? bf2f(residbf[row * HID + h])
                           : U[ro + 3 * HID + h];
        float f  = 1.0f / (1.0f + __expf(-(u1 + vf * c + bf_)));
        c = f * c + (1.0f - f) * u0;
        float rg = 1.0f / (1.0f + __expf(-(u2 + vr * c + br)));
        float hv = rg * tanhf(c) + (1.0f - rg) * xr * SCALE_X;
        if (hout) hout[row * HID + h] = f2bf(hv);
        hsum += hv;
    }
    if (hbar) hbar[g] = hsum * (1.0f / (float)T_SZ);
}

// ---------------------------------------------------------------------------
// out[b,c] = hbar[b,:] . fc_w[:,c] + fc_b[c]   (hbar already T-averaged)
// ---------------------------------------------------------------------------
__global__ void fc_kernel(const float* __restrict__ hbar,
                          const float* __restrict__ fcw,
                          const float* __restrict__ fcb,
                          float* __restrict__ out)
{
    int b = blockIdx.x;
    int c = threadIdx.x;
    if (c >= NCLS) return;
    float acc = fcb[c];
    const float* hb = hbar + (size_t)b * HID;
    for (int h = 0; h < HID; ++h)
        acc += hb[h] * fcw[(size_t)h * NCLS + c];
    out[(size_t)b * NCLS + c] = acc;
}

// ---------------------------------------------------------------------------
extern "C" void kernel_launch(void* const* d_in, const int* in_sizes, int n_in,
                              void* d_out, int out_size, void* d_ws, size_t ws_size,
                              hipStream_t stream)
{
    (void)in_sizes; (void)n_in; (void)out_size; (void)ws_size;
    const float* x   = (const float*)d_in[0];
    const float* W0  = (const float*)d_in[1];
    const float* W1  = (const float*)d_in[2];
    const float* W2  = (const float*)d_in[3];
    const float* vc0 = (const float*)d_in[4];
    const float* vc1 = (const float*)d_in[5];
    const float* vc2 = (const float*)d_in[6];
    const float* b0  = (const float*)d_in[7];
    const float* b1  = (const float*)d_in[8];
    const float* b2  = (const float*)d_in[9];
    const float* fcw = (const float*)d_in[10];
    const float* fcb = (const float*)d_in[11];
    float* out = (float*)d_out;

    char* ws = (char*)d_ws;
    size_t off = 0;
    auto alloc = [&](size_t bytes) -> char* {
        char* p = ws + off;
        off += (bytes + 255) & ~(size_t)255;
        return p;
    };
    unsigned short* xbf  = (unsigned short*)alloc((size_t)M_ROWS * IMG * 2);   // 25 MB
    unsigned short* wt   = (unsigned short*)alloc((size_t)4096 * 1024 * 2);    //  8 MB
    float*          U    = (float*)alloc((size_t)M_ROWS * 4096 * 4);           // 403 MB
    unsigned short* hA   = (unsigned short*)alloc((size_t)M_ROWS * HID * 2);   // 50 MB
    unsigned short* hB   = (unsigned short*)alloc((size_t)M_ROWS * HID * 2);   // 50 MB
    float*          hbar = (float*)alloc((size_t)B_SZ * HID * 4);              //  8 MB

    const int scan_blocks = (B_SZ * HID) / 256;

    // ---- Layer 0: k=4, K=IMG, N=4*HID; resid comes from U column block 3.
    cvt_x_kernel<<<(M_ROWS * IMG) / 256, 256, 0, stream>>>(x, xbf);
    cvt_w_kernel<<<(IMG * 4 * HID) / 256, 256, 0, stream>>>(W0, wt, IMG, 4 * HID);
    {
        dim3 grid((4 * HID) / 128, M_ROWS / 128);
        gemm_bf16_kernel<<<grid, 256, 0, stream>>>(xbf, wt, U, M_ROWS, 4 * HID, IMG);
    }
    sru_scan_kernel<<<scan_blocks, 256, 0, stream>>>(U, 4 * HID, nullptr,
                                                     vc0, b0, hA, nullptr);

    // ---- Layer 1: k=3, K=HID, N=3*HID; resid = layer-0 output hA.
    cvt_w_kernel<<<(HID * 3 * HID) / 256, 256, 0, stream>>>(W1, wt, HID, 3 * HID);
    {
        dim3 grid((3 * HID) / 128, M_ROWS / 128);
        gemm_bf16_kernel<<<grid, 256, 0, stream>>>(hA, wt, U, M_ROWS, 3 * HID, HID);
    }
    sru_scan_kernel<<<scan_blocks, 256, 0, stream>>>(U, 3 * HID, hA,
                                                     vc1, b1, hB, nullptr);

    // ---- Layer 2: resid = hB; emit only the T-mean (classifier mean is linear).
    cvt_w_kernel<<<(HID * 3 * HID) / 256, 256, 0, stream>>>(W2, wt, HID, 3 * HID);
    {
        dim3 grid((3 * HID) / 128, M_ROWS / 128);
        gemm_bf16_kernel<<<grid, 256, 0, stream>>>(hB, wt, U, M_ROWS, 3 * HID, HID);
    }
    sru_scan_kernel<<<scan_blocks, 256, 0, stream>>>(U, 3 * HID, hB,
                                                     vc2, b2, nullptr, hbar);

    // ---- Classifier on T-averaged hidden state.
    fc_kernel<<<B_SZ, 64, 0, stream>>>(hbar, fcw, fcb, out);
}
